// MHA_26087631356396
// MI455X (gfx1250) — matrix operand order repaired and checked by reading, not compiled, not run on hardware
//
#include <hip/hip_runtime.h>
#include <hip/hip_bf16.h>
#include <math.h>

typedef __attribute__((ext_vector_type(16))) _Float16 v16h;
typedef __attribute__((ext_vector_type(8)))  float    v8f;

#define HEADS 12
#define B_ 8
#define L_ 512
#define E_ 768
#define DH 64
#define FF 64
#define EPS_ 1e-6f
#define QK_SCALE 0.35355339059327373f   /* 64^-0.25 */
#define C1 0.125f                        /* 64^-0.5  */

// ---------------------------------------------------------------------------
// Fragment loaders (wave32); A and B both use contiguous k order K = 16*(lane/16) + e.
// A (16x32 f16, MxK):  lane<16: m=lane,  K=e
//                      lane>=16: m=lane-16, K=16+e
// B (32x16 f16, KxN):  lane<16: n=lane,  K=e ;  lane>=16: n=lane-16, K=16+e
// C/D (16x16 f32):     vgpr r: lanes0-15 M=r, N=lane; lanes16-31 M=8+r
// A tile stored row-major [m][k]; B tile stored TRANSPOSED [n][k] so each
// lane's 16 fragment halves are contiguous -> 2x ds_load_b128 per fragment.
// ---------------------------------------------------------------------------
__device__ __forceinline__ v16h load_a_frag(const _Float16* Ah, int pitch,
                                            int m0, int lane) {
  v16h a;
  const _Float16* row = Ah + (m0 + (lane & 15)) * pitch;
  const int koff = (lane >> 4) << 4;
#pragma unroll
  for (int e = 0; e < 8; ++e) a[e] = row[koff + e];
#pragma unroll
  for (int e = 0; e < 8; ++e) a[8 + e] = row[8 + koff + e];
  return a;
}

__device__ __forceinline__ v16h load_bt_frag(const _Float16* Bt, int pitch,
                                             int n0, int lane) {
  v16h b;
  const _Float16* p = Bt + (n0 + (lane & 15)) * pitch + (lane & 16);
#pragma unroll
  for (int e = 0; e < 16; ++e) b[e] = p[e];
  return b;
}

// ---------------------------------------------------------------------------
// Kernel 1/6: C[M,N] = A[M,K] @ B[K,N] + bias, fp32 in/out, f16 WMMA compute.
// Block tile 128x128, 256 threads = 8 waves in 4x2, wave tile 32x64 (2x4 WMMA).
// N and K are compile-time constants so all staging strides fold into
// immediate-offset global loads (clause-able, no per-element 64-bit adds).
// Requires M%128==0, N%128==0, K%32==0 (true for all uses here).
// ---------------------------------------------------------------------------
template <int N, int K>
__global__ __launch_bounds__(256)
void gemm_wmma_kernel(const float* __restrict__ A, const float* __restrict__ Bm,
                      const float* __restrict__ bias, float* __restrict__ C) {
  __shared__ _Float16 Ah[128 * 40];   // [m][k], pitch 40 halves (80B rows)
  __shared__ _Float16 Bt[128 * 40];   // [n][k] TRANSPOSED, pitch 40 halves

  const int tid  = threadIdx.x;
  const int lane = tid & 31;
  const int wave = tid >> 5;
  const int wr = wave >> 1;           // 0..3  (row of 32)
  const int wc = wave & 1;            // 0..1  (col of 64)
  const int bm = blockIdx.x * 128;
  const int bn = blockIdx.y * 128;

  v8f acc[2][4] = {};

  const int arow = tid >> 1, acol = (tid & 1) * 16;   // A: 16 fp32/thread
  const int bnl  = tid >> 1, bkh  = (tid & 1) * 16;   // B: column n, 16 k's

  const float* ap_base = A + (size_t)(bm + arow) * K + acol;
  const float* bp_base = Bm + (size_t)bkh * N + bn + bnl;

  for (int kk = 0; kk < K; kk += 32) {
    // stage A 128x32 fp32 -> f16 LDS (row-major, contiguous halves)
    {
      const float4* ap = (const float4*)(ap_base + kk);
      _Float16* ad = Ah + arow * 40 + acol;
#pragma unroll
      for (int q = 0; q < 4; ++q) {
        float4 v = ap[q];
        ad[q * 4 + 0] = (_Float16)v.x; ad[q * 4 + 1] = (_Float16)v.y;
        ad[q * 4 + 2] = (_Float16)v.z; ad[q * 4 + 3] = (_Float16)v.w;
      }
      if (kk + 32 < K)  // next K-tile -> global_prefetch_b8
        __builtin_prefetch(ap_base + kk + 32, 0, 1);
    }
    // stage B 32x128 fp32 -> f16 LDS transposed: Bt[n][k]
    // 16 dword loads from ONE base vaddr at constant j*N immediate offsets
    // (coalesced across lanes at each j; j*N*4 <= 138KB < 2^23).
    {
      const float* bp = bp_base + (size_t)kk * N;
      float vals[16];
#pragma unroll
      for (int j = 0; j < 16; ++j) vals[j] = bp[j * N];
      _Float16* bd = Bt + bnl * 40 + bkh;
#pragma unroll
      for (int j = 0; j < 16; ++j) bd[j] = (_Float16)vals[j];
      if (kk + 32 < K)
        __builtin_prefetch(bp + 32 * N, 0, 1);
    }
    __syncthreads();

    v16h af[2];
    af[0] = load_a_frag(Ah, 40, wr * 32, lane);
    af[1] = load_a_frag(Ah, 40, wr * 32 + 16, lane);
    v16h bf[4];
#pragma unroll
    for (int j = 0; j < 4; ++j)
      bf[j] = load_bt_frag(Bt, 40, wc * 64 + j * 16, lane);

#pragma unroll
    for (int i = 0; i < 2; ++i)
#pragma unroll
      for (int j = 0; j < 4; ++j)
        acc[i][j] = __builtin_amdgcn_wmma_f32_16x16x32_f16(
            false, af[i], false, bf[j], (short)0, acc[i][j], false, false);
    __syncthreads();
  }

  // write back with bias (C/D layout: vgpr r -> row (lane>>4)*8 + r, col lane&15)
#pragma unroll
  for (int i = 0; i < 2; ++i)
#pragma unroll
    for (int j = 0; j < 4; ++j) {
      const int col = bn + wc * 64 + j * 16 + (lane & 15);
      const float bv = bias[col];
      const int rbase = bm + wr * 32 + i * 16 + ((lane >> 4) << 3);
#pragma unroll
      for (int r = 0; r < 8; ++r)
        C[(size_t)(rbase + r) * N + col] = acc[i][j][r] + bv;
    }
}

// ---------------------------------------------------------------------------
// Kernel 2/6: FAVOR feature projection s = (seq * d^-0.25) @ omega^T + stats.
// grid = (B*H, L/128, 2)  z: 0=query, 1=key.  8 waves x 16 rows, N=64, K=64.
// omega staged in native [f][d] layout == transposed-B layout for fragments.
// Query: finished in-kernel (row max). Key: raw s + diag + rowmax to ws.
// ---------------------------------------------------------------------------
__global__ __launch_bounds__(256)
void favor_features_kernel(const float* __restrict__ c_buf,
                           const float* __restrict__ omega,
                           float* __restrict__ qf, float* __restrict__ sk_raw,
                           float* __restrict__ diag_k, float* __restrict__ rowmax_k) {
  __shared__ _Float16 Ot[64 * 72];          // Ot[f][d] = omega[f][d], pitch 72
  __shared__ float Sf[128 * 65];            // s tile, pitch 65
  __shared__ float psum[256], pmax[256];
  __shared__ float rdiag[128], rmax_s[128];

  const int tid = threadIdx.x, lane = tid & 31, wave = tid >> 5;
  const int bh = blockIdx.x;
  const int b = bh / HEADS, h = bh % HEADS;
  const int l0b = blockIdx.y * 128;
  const int isK = blockIdx.z;

  for (int i = tid; i < 64 * 64; i += 256) {   // flat [f][d] copy
    const int f = i >> 6, d = i & 63;
    Ot[f * 72 + d] = (_Float16)omega[i];
  }
  __syncthreads();

  const int lw = l0b + wave * 16;
  const int coff = (isK ? E_ : 0) + h * DH;
  const float* rowp =
      c_buf + (size_t)(b * L_ + lw + (lane & 15)) * (3 * E_) + coff;

  v8f acc[4] = {};
#pragma unroll
  for (int s = 0; s < 2; ++s) {
    v16h a;
#pragma unroll
    for (int e = 0; e < 16; ++e) {
      const int k = s * 32 + ((lane >> 4) << 4) + e;
      a[e] = (_Float16)(rowp[k] * QK_SCALE);
    }
#pragma unroll
    for (int j = 0; j < 4; ++j) {
      v16h bf;
      const _Float16* p = Ot + (j * 16 + (lane & 15)) * 72 + s * 32 + (lane & 16);
#pragma unroll
      for (int e = 0; e < 16; ++e) bf[e] = p[e];
      acc[j] = __builtin_amdgcn_wmma_f32_16x16x32_f16(false, a, false, bf,
                                                      (short)0, acc[j], false, false);
    }
  }
#pragma unroll
  for (int j = 0; j < 4; ++j) {
    const int col = j * 16 + (lane & 15);
    const int rb = wave * 16 + ((lane >> 4) << 3);
#pragma unroll
    for (int r = 0; r < 8; ++r) Sf[(rb + r) * 65 + col] = acc[j][r];
  }
  __syncthreads();

  {  // row sum + max, 2 threads/row
    const int row = tid >> 1, half = tid & 1;
    const float* sp = Sf + row * 65 + half * 32;
    float s = 0.0f, mx = -3.4e38f;
#pragma unroll
    for (int i = 0; i < 32; ++i) { float v = sp[i]; s += v; mx = fmaxf(mx, v); }
    psum[tid] = s; pmax[tid] = mx;
  }
  __syncthreads();
  if (tid < 128) {
    const float s  = psum[2 * tid] + psum[2 * tid + 1];
    const float mx = fmaxf(pmax[2 * tid], pmax[2 * tid + 1]);
    rdiag[tid] = 0.5f * s;
    rmax_s[tid] = mx;
    if (isK) {
      const size_t idx = (size_t)bh * L_ + (l0b + tid);
      diag_k[idx] = 0.5f * s;
      rowmax_k[idx] = mx;
    }
  }
  __syncthreads();

  for (int i = tid; i < 128 * 64; i += 256) {
    const int row = i >> 6, f = i & 63;
    const float sv = Sf[row * 65 + f];
    const size_t oidx = ((size_t)bh * L_ + (l0b + row)) * FF + f;
    if (isK) sk_raw[oidx] = sv;  // finished later (needs cross-head max)
    else     qf[oidx] = C1 * (expf(sv - rdiag[row] - rmax_s[row]) + EPS_);
  }
}

// Kernel 3/6: maxk[b,l] = max over heads of per-row key max.
__global__ void max_over_heads_kernel(const float* __restrict__ rowmax_k,
                                      float* __restrict__ maxk) {
  const int i = blockIdx.x * 256 + threadIdx.x;
  if (i >= B_ * L_) return;
  const int b = i / L_, l = i % L_;
  float m = -3.4e38f;
#pragma unroll
  for (int h = 0; h < HEADS; ++h)
    m = fmaxf(m, rowmax_k[((size_t)(b * HEADS + h)) * L_ + l]);
  maxk[i] = m;
}

// Kernel 4/6: kf = F^-0.5 * (exp(s - diag - maxk) + eps), in place on raw s.
__global__ void kf_finalize_kernel(float* __restrict__ kf,
                                   const float* __restrict__ diag_k,
                                   const float* __restrict__ maxk) {
  const size_t i = (size_t)blockIdx.x * 256 + threadIdx.x;
  const size_t total = (size_t)B_ * HEADS * L_ * FF;
  if (i >= total) return;
  const size_t r = i >> 6;               // (b,h,l)
  const int l = (int)(r % L_);
  const int b = (int)(r / L_) / HEADS;
  kf[i] = C1 * (expf(kf[i] - diag_k[r] - maxk[b * L_ + l]) + EPS_);
}

// ---------------------------------------------------------------------------
// Kernel 5/6: causal linear-attention scan. One workgroup per (b,h).
// S[64][64] state in LDS with pitch 65 (conflict-free); sequential over L.
// ---------------------------------------------------------------------------
__global__ __launch_bounds__(256)
void causal_scan_kernel(const float* __restrict__ qf, const float* __restrict__ kf,
                        const float* __restrict__ c_buf, float* __restrict__ a_buf) {
  __shared__ float S[64 * 65];
  __shared__ float z[64];
  __shared__ float kf_s[64], v_s[64], qf_s[64];
  __shared__ float part[64 * 4];
  __shared__ float dpart[16];

  const int tid = threadIdx.x;
  const int bh = blockIdx.x;
  const int b = bh / HEADS, h = bh % HEADS;

  for (int i = tid; i < 64 * 65; i += 256) S[i] = 0.0f;
  if (tid < 64) z[tid] = 0.0f;
  __syncthreads();

  const int d = tid >> 2;
  const int fb = (tid & 3) << 4;

  for (int l = 0; l < L_; ++l) {
    if (tid < 64) {
      const size_t fidx = ((size_t)bh * L_ + l) * FF + tid;
      kf_s[tid] = kf[fidx];
      qf_s[tid] = qf[fidx];
      v_s[tid]  = c_buf[(size_t)(b * L_ + l) * (3 * E_) + 2 * E_ + h * DH + tid];
    }
    __syncthreads();
    {
      const float vd = v_s[d];
      float p = 0.0f;
      float* Srow = S + d * 65 + fb;
#pragma unroll
      for (int j = 0; j < 16; ++j) {
        const float sv = Srow[j] + vd * kf_s[fb + j];
        Srow[j] = sv;
        p += sv * qf_s[fb + j];
      }
      part[d * 4 + (tid & 3)] = p;
    }
    if (tid < 16) {  // z cumsum + denom partials (disjoint f per thread)
      float zp = 0.0f;
#pragma unroll
      for (int j = 0; j < 4; ++j) {
        const int f = tid * 4 + j;
        const float zv = z[f] + kf_s[f];
        z[f] = zv;
        zp += zv * qf_s[f];
      }
      dpart[tid] = zp;
    }
    __syncthreads();
    if (tid < 64) {
      const float num = part[tid * 4] + part[tid * 4 + 1] +
                        part[tid * 4 + 2] + part[tid * 4 + 3];
      float den = 0.0f;
#pragma unroll
      for (int j = 0; j < 16; ++j) den += dpart[j];
      a_buf[(size_t)(b * L_ + l) * E_ + h * DH + tid] = num / (den + EPS_);
    }
    __syncthreads();
  }
}

// ---------------------------------------------------------------------------
extern "C" void kernel_launch(void* const* d_in, const int* in_sizes, int n_in,
                              void* d_out, int out_size, void* d_ws, size_t ws_size,
                              hipStream_t stream) {
  (void)in_sizes; (void)n_in; (void)out_size; (void)ws_size;
  const float* x     = (const float*)d_in[0];
  const float* w_inp = (const float*)d_in[1];
  const float* b_inp = (const float*)d_in[2];
  const float* w_out = (const float*)d_in[3];
  const float* b_out = (const float*)d_in[4];
  const float* omega = (const float*)d_in[5];
  float* out = (float*)d_out;

  float* ws = (float*)d_ws;
  float* c_buf    = ws;                    // B*L*3E   = 9,437,184
  float* qf       = c_buf + 9437184;       // B*H*L*F  = 3,145,728
  float* kf       = qf + 3145728;          // B*H*L*F  = 3,145,728 (raw s -> kf)
  float* diag_k   = kf + 3145728;          // B*H*L    = 49,152
  float* rowmax_k = diag_k + 49152;        // B*H*L    = 49,152
  float* maxk     = rowmax_k + 49152;      // B*L      = 4,096
  float* a_buf    = maxk + 4096;           // B*L*E    = 3,145,728

  const dim3 blk(256);
  // 1) fused QKV projection: [4096,768] @ [768,2304] + b_inp
  gemm_wmma_kernel<3 * E_, E_><<<dim3(4096 / 128, 2304 / 128), blk, 0, stream>>>(
      x, w_inp, b_inp, c_buf);
  // 2) FAVOR feature maps for q and k
  favor_features_kernel<<<dim3(B_ * HEADS, L_ / 128, 2), blk, 0, stream>>>(
      c_buf, omega, qf, kf, diag_k, rowmax_k);
  // 3) key max over (heads, features)
  max_over_heads_kernel<<<dim3((B_ * L_ + 255) / 256), blk, 0, stream>>>(
      rowmax_k, maxk);
  // 4) finalize kf
  kf_finalize_kernel<<<dim3((int)((3145728 + 255) / 256)), blk, 0, stream>>>(
      kf, diag_k, maxk);
  // 5) causal linear-attention prefix scan
  causal_scan_kernel<<<dim3(B_ * HEADS), blk, 0, stream>>>(qf, kf, c_buf, a_buf);
  // 6) output projection: [4096,768] @ [768,768] + b_out
  gemm_wmma_kernel<E_, E_><<<dim3(4096 / 128, 768 / 128), blk, 0, stream>>>(
      a_buf, w_out, b_out, out);
}
